// NormLinearAttention_120259084870
// MI455X (gfx1250) — matrix operand order repaired and checked by reading, not compiled, not run on hardware
//
#include <hip/hip_runtime.h>

typedef __attribute__((ext_vector_type(16))) __bf16 v16bf;
typedef __attribute__((ext_vector_type(8)))  float  v8f;
typedef __attribute__((ext_vector_type(4)))  int    v4i;

#define D_   1024
#define BN   32768      // B*N rows
#define NSEQ 8192
#define HEADS 16
#define HD   64
#define LDA  40         // padded LDS row stride in bf16 (80B -> conflict-free b128 reads)

#if defined(__has_builtin)
#if __has_builtin(__builtin_amdgcn_global_load_async_to_lds_b128)
#define HAVE_ASYNC_B128 1
#endif
#if __has_builtin(__builtin_amdgcn_s_wait_asynccnt)
#define HAVE_WAIT_ASYNC 1
#endif
#endif

typedef __attribute__((address_space(1))) v4i* gv4i_p;   // global int4*
typedef __attribute__((address_space(3))) v4i* lv4i_p;   // LDS int4*

union FragU { uint4 u[2]; v16bf v; };

static __device__ __forceinline__ v16bf make_frag(uint4 a, uint4 b) {
  FragU f; f.u[0] = a; f.u[1] = b; return f.v;
}

static __device__ __forceinline__ v8f wmma_bf16(v16bf a, v16bf b, v8f c) {
  // emits v_wmma_f32_16x16x32_bf16
  return __builtin_amdgcn_wmma_f32_16x16x32_bf16(false, a, false, b, (short)0, c, false, false);
}

// 16B global -> LDS copy; async (ASYNCcnt-tracked) when the CDNA5 builtin exists
static __device__ __forceinline__ void cp16_to_lds(__bf16* l, const __bf16* g) {
#if defined(HAVE_ASYNC_B128)
  __builtin_amdgcn_global_load_async_to_lds_b128((gv4i_p)g, (lv4i_p)l, 0, 0);
#else
  *(uint4*)l = *(const uint4*)g;
#endif
}

static __device__ __forceinline__ void wait_async0() {
#if defined(HAVE_ASYNC_B128)
#if defined(HAVE_WAIT_ASYNC)
  __builtin_amdgcn_s_wait_asynccnt(0);
#else
  asm volatile("s_wait_asynccnt 0x0" ::: "memory");
#endif
#endif
}

// A-fragment: 16x32 bf16 tile (lane&15 = M; lane half selects K 0-7/16-23 vs 8-15/24-31)
static __device__ __forceinline__ v16bf load_a(const __bf16* p0, int stride, int lane) {
  const __bf16* p = p0 + (long)(lane & 15) * stride + ((lane >> 4) << 3);
  uint4 x = *(const uint4*)p;
  uint4 y = *(const uint4*)(p + 16);
  return make_frag(x, y);
}

// B-fragment: 32x16 tile from N-major (transposed) storage (lane&15 = N; half selects K 0-7/16-23 vs 8-15/24-31)
static __device__ __forceinline__ v16bf load_b(const __bf16* p0, int stride, int lane) {
  const __bf16* p = p0 + (long)(lane & 15) * stride + ((lane >> 4) << 3);
  uint4 x = *(const uint4*)p;
  uint4 y = *(const uint4*)(p + 16);
  return make_frag(x, y);
}

// ---------------- conversions ----------------
__global__ __launch_bounds__(256)
void cvt_bf16(const float* __restrict__ src, __bf16* __restrict__ dst, long n) {
  long i = ((long)blockIdx.x * 256 + threadIdx.x) * 4;
  if (i >= n) return;
  float4 v = *(const float4*)(src + i);
  dst[i + 0] = (__bf16)v.x; dst[i + 1] = (__bf16)v.y;
  dst[i + 2] = (__bf16)v.z; dst[i + 3] = (__bf16)v.w;
}

// Wt[n][k] = W[k][n]  (coalesced read, scattered bf16 write; tiny, one-shot)
__global__ __launch_bounds__(256)
void cvt_wt(const float* __restrict__ W, __bf16* __restrict__ Wt) {
  int i = blockIdx.x * 256 + threadIdx.x;   // 0 .. 1M-1
  int k = i >> 10, n = i & 1023;
  Wt[(long)n * D_ + k] = (__bf16)W[(long)k * D_ + n];
}

// ---------------- generic 128x128 bf16 GEMM with epilogue modes ----------------
// mode 0: bf16 row-major out, ReLU        (q)
// mode 1: bf16 head-transposed kT, ReLU   (k)
// mode 2: bf16 head-transposed vT         (v)
// mode 3: fp32 row-major out              (u / final)
__global__ __launch_bounds__(256)
void gemm128(const __bf16* __restrict__ A, const __bf16* __restrict__ Bt,
             const float* __restrict__ bias, void* __restrict__ out,
             int M, int K, int N, int mode) {
  __shared__ __bf16 As[2][128 * LDA];
  __shared__ __bf16 Bs[2][128 * LDA];
  int t = threadIdx.x;
  int lane = t & 31;
  int wv = t >> 5;
  int wm = wv >> 2;      // 0..1 : 64-row strip
  int wn = wv & 3;       // 0..3 : 32-col strip
  long blockM = (long)blockIdx.y * 128;
  long blockN = (long)blockIdx.x * 128;

  // staging: each thread moves 2x16B for A and B per tile (128x32 bf16 = 8KB each)
  int id0 = t, id1 = t + 256;
  int ar0 = id0 >> 2, ac0 = (id0 & 3) << 3;
  int ar1 = id1 >> 2, ac1 = (id1 & 3) << 3;

  const __bf16* Ag = A + blockM * K;
  const __bf16* Bg = Bt + blockN * K;

  // stage tile 0 (async to LDS on CDNA5)
  cp16_to_lds(&As[0][ar0 * LDA + ac0], Ag + (long)ar0 * K + ac0);
  cp16_to_lds(&As[0][ar1 * LDA + ac1], Ag + (long)ar1 * K + ac1);
  cp16_to_lds(&Bs[0][ar0 * LDA + ac0], Bg + (long)ar0 * K + ac0);
  cp16_to_lds(&Bs[0][ar1 * LDA + ac1], Bg + (long)ar1 * K + ac1);
  wait_async0();
  __syncthreads();

  v8f acc[4][2] = {};
  int KT = K >> 5;
  for (int kt = 0; kt < KT; ++kt) {
    int cur = kt & 1;
    if (kt + 1 < KT) {  // issue async staging of next tile into the other buffer
      int k0 = (kt + 1) << 5;
      int nxt = cur ^ 1;
      cp16_to_lds(&As[nxt][ar0 * LDA + ac0], Ag + (long)ar0 * K + k0 + ac0);
      cp16_to_lds(&As[nxt][ar1 * LDA + ac1], Ag + (long)ar1 * K + k0 + ac1);
      cp16_to_lds(&Bs[nxt][ar0 * LDA + ac0], Bg + (long)ar0 * K + k0 + ac0);
      cp16_to_lds(&Bs[nxt][ar1 * LDA + ac1], Bg + (long)ar1 * K + k0 + ac1);
    }
    const __bf16* as = As[cur];
    const __bf16* bs = Bs[cur];
    v16bf af[4], bf2[2];
#pragma unroll
    for (int mt = 0; mt < 4; ++mt)
      af[mt] = load_a(as + (wm * 64 + mt * 16) * LDA, LDA, lane);
#pragma unroll
    for (int nt = 0; nt < 2; ++nt)
      bf2[nt] = load_b(bs + (wn * 32 + nt * 16) * LDA, LDA, lane);
#pragma unroll
    for (int mt = 0; mt < 4; ++mt)
#pragma unroll
      for (int nt = 0; nt < 2; ++nt)
        acc[mt][nt] = wmma_bf16(af[mt], bf2[nt], acc[mt][nt]);
    if (kt + 1 < KT) {
      wait_async0();     // our async writes to buf[nxt] are done
      __syncthreads();   // everyone finished compute(kt) and staging(kt+1)
    }
  }

#pragma unroll
  for (int mt = 0; mt < 4; ++mt)
#pragma unroll
    for (int nt = 0; nt < 2; ++nt)
#pragma unroll
      for (int r = 0; r < 8; ++r) {
        long row = blockM + wm * 64 + mt * 16 + r + ((lane >> 4) << 3);
        int  col = (int)(blockN + wn * 32 + nt * 16 + (lane & 15));
        float v = acc[mt][nt][r] + bias[col];
        if (mode == 0) {
          ((__bf16*)out)[row * N + col] = (__bf16)fmaxf(v, 0.0f);
        } else if (mode <= 2) {
          if (mode == 1) v = fmaxf(v, 0.0f);
          long b = row >> 13, n = row & 8191;
          int  h = col >> 6, dd = col & 63;
          ((__bf16*)out)[(((b * HEADS + h) * HD + dd) << 13) + n] = (__bf16)v;
        } else {
          ((float*)out)[row * N + col] = v;
        }
      }
}

// ---------------- kv = sum_n k^T v (split-K partials) ----------------
__global__ __launch_bounds__(256)
void kv_partial(const __bf16* __restrict__ kT, const __bf16* __restrict__ vT,
                float* __restrict__ part) {
  int bh = blockIdx.x;          // 0..63
  int sp = blockIdx.y;          // 0..7 split
  int t = threadIdx.x, lane = t & 31, wv = t >> 5;
  int mt = wv & 3;              // d-tile
  int ntb = (wv >> 2) << 1;     // e-tile base (0 or 2)
  const __bf16* Ab = kT + (long)bh * HD * NSEQ + (long)sp * 1024;
  const __bf16* Bb = vT + (long)bh * HD * NSEQ + (long)sp * 1024;
  v8f acc0 = {}, acc1 = {};
  for (int ks = 0; ks < 1024; ks += 32) {
    if (ks + 64 < 1024) {  // pull next chunks toward cache (global_prefetch_b8)
      __builtin_prefetch(Ab + (long)(mt * 16 + (lane & 15)) * NSEQ + ks + 64, 0, 1);
      __builtin_prefetch(Bb + (long)(ntb * 16 + (lane & 15)) * NSEQ + ks + 64, 0, 1);
    }
    v16bf a  = load_a(Ab + (long)(mt * 16) * NSEQ + ks, NSEQ, lane);
    v16bf b0 = load_b(Bb + (long)(ntb * 16) * NSEQ + ks, NSEQ, lane);
    v16bf b1 = load_b(Bb + (long)((ntb + 1) * 16) * NSEQ + ks, NSEQ, lane);
    acc0 = wmma_bf16(a, b0, acc0);
    acc1 = wmma_bf16(a, b1, acc1);
  }
  float* pp = part + ((long)bh * 8 + sp) * 4096;
#pragma unroll
  for (int r = 0; r < 8; ++r) {
    int d  = mt * 16 + r + ((lane >> 4) << 3);
    int e0 = ntb * 16 + (lane & 15);
    pp[d * 64 + e0]      = acc0[r];
    pp[d * 64 + e0 + 16] = acc1[r];
  }
}

// reduce splits + sign-preserving abs-clamp; store transposed [e][d] as bf16
__global__ __launch_bounds__(256)
void kv_reduce(const float* __restrict__ part, __bf16* __restrict__ kvT) {
  int idx = blockIdx.x * 256 + threadIdx.x;     // 64*4096
  int bh = idx >> 12, rem = idx & 4095;
  int d = rem >> 6, e = rem & 63;
  float s = 0.0f;
#pragma unroll
  for (int sp = 0; sp < 8; ++sp) s += part[((long)bh * 8 + sp) * 4096 + rem];
  float a = fminf(fmaxf(fabsf(s), 0.01f), 100.0f);
  float v = s > 0.0f ? a : (s < 0.0f ? -a : 0.0f);
  kvT[(long)bh * 4096 + e * 64 + d] = (__bf16)v;
}

// ---------------- out = q @ kv per (b,h) ----------------
__global__ __launch_bounds__(256)
void attn_out(const __bf16* __restrict__ q, const __bf16* __restrict__ kvT,
              float* __restrict__ attn) {
  int bh = blockIdx.y; int b = bh >> 4; int h = bh & 15;
  long m0 = (long)blockIdx.x * 128;
  int t = threadIdx.x, lane = t & 31, wv = t >> 5;
  const __bf16* Ab = q + (long)b * NSEQ * D_ + h * HD;
  const __bf16* Bb = kvT + (long)bh * 4096;
  v8f acc[4] = {};
#pragma unroll
  for (int ks = 0; ks < 64; ks += 32) {
    v16bf a = load_a(Ab + (m0 + wv * 16) * D_ + ks, D_, lane);
#pragma unroll
    for (int nt = 0; nt < 4; ++nt) {
      v16bf bb = load_b(Bb + (nt * 16) * HD + ks, HD, lane);
      acc[nt] = wmma_bf16(a, bb, acc[nt]);
    }
  }
#pragma unroll
  for (int nt = 0; nt < 4; ++nt)
#pragma unroll
    for (int r = 0; r < 8; ++r) {
      long n = m0 + wv * 16 + r + ((lane >> 4) << 3);
      int  e = nt * 16 + (lane & 15);
      attn[((long)b * NSEQ + n) * D_ + h * HD + e] = acc[nt][r];
    }
}

// ---------------- LayerNorm(out) * u -> bf16 ----------------
__global__ __launch_bounds__(256)
void ln_mul(const float* __restrict__ attn, const float* __restrict__ u,
            const float* __restrict__ lnw, const float* __restrict__ lnb,
            __bf16* __restrict__ x2) {
  __shared__ float s1[256], s2[256];
  long row = blockIdx.x;
  int t = threadIdx.x;
  float4 v = ((const float4*)(attn + row * D_))[t];
  s1[t] = v.x + v.y + v.z + v.w;
  s2[t] = v.x * v.x + v.y * v.y + v.z * v.z + v.w * v.w;
  __syncthreads();
  for (int o = 128; o > 0; o >>= 1) {
    if (t < o) { s1[t] += s1[t + o]; s2[t] += s2[t + o]; }
    __syncthreads();
  }
  float mu  = s1[0] * (1.0f / D_);
  float var = s2[0] * (1.0f / D_) - mu * mu;
  float rs  = rsqrtf(var + 1e-5f);
  float4 uu = ((const float4*)(u + row * D_))[t];
  float4 w  = ((const float4*)lnw)[t];
  float4 bb = ((const float4*)lnb)[t];
  __bf16* o2 = x2 + row * D_ + t * 4;
  o2[0] = (__bf16)(((v.x - mu) * rs * w.x + bb.x) * uu.x);
  o2[1] = (__bf16)(((v.y - mu) * rs * w.y + bb.y) * uu.y);
  o2[2] = (__bf16)(((v.z - mu) * rs * w.z + bb.z) * uu.z);
  o2[3] = (__bf16)(((v.w - mu) * rs * w.w + bb.w) * uu.w);
}

extern "C" void kernel_launch(void* const* d_in, const int* in_sizes, int n_in,
                              void* d_out, int out_size, void* d_ws, size_t ws_size,
                              hipStream_t stream) {
  (void)in_sizes; (void)n_in; (void)out_size; (void)ws_size;
  const float* query = (const float*)d_in[0];
  const float* Wq = (const float*)d_in[1];  const float* bq = (const float*)d_in[2];
  const float* Wk = (const float*)d_in[3];  const float* bk = (const float*)d_in[4];
  const float* Wv = (const float*)d_in[5];  const float* bv = (const float*)d_in[6];
  const float* Wu = (const float*)d_in[7];  const float* bu = (const float*)d_in[8];
  const float* Wo = (const float*)d_in[9];  const float* bo = (const float*)d_in[10];
  const float* lnw = (const float*)d_in[11]; const float* lnb = (const float*)d_in[12];

  char* ws = (char*)d_ws;
  size_t off = 0;
  __bf16* qry_bf = (__bf16*)(ws + off); off += (size_t)BN * D_ * 2;          // 64MB (reused as x2)
  __bf16* wt     = (__bf16*)(ws + off); off += (size_t)5 * D_ * D_ * 2;      // 10MB
  __bf16* q_bf   = (__bf16*)(ws + off); off += (size_t)BN * D_ * 2;          // 64MB
  __bf16* kT     = (__bf16*)(ws + off); off += (size_t)BN * D_ * 2;          // 64MB
  __bf16* vT     = (__bf16*)(ws + off); off += (size_t)BN * D_ * 2;          // 64MB
  float*  u      = (float*) (ws + off); off += (size_t)BN * D_ * 4;          // 128MB
  float*  kvp    = (float*) (ws + off); off += (size_t)64 * 8 * 4096 * 4;    // 8MB
  __bf16* kvT    = (__bf16*)(ws + off); off += (size_t)64 * 4096 * 2;        // 0.5MB
  float*  attn   = (float*) (ws + off); off += (size_t)BN * D_ * 4;          // 128MB
  __bf16* x2     = qry_bf;  // alias: query_bf16 fully consumed before ln_mul writes

  const size_t WSZ = (size_t)D_ * D_;
  cvt_bf16<<<(BN * (D_ / 4)) / 256, 256, 0, stream>>>(query, qry_bf, (long)BN * D_);
  cvt_wt<<<4096, 256, 0, stream>>>(Wq, wt + 0 * WSZ);
  cvt_wt<<<4096, 256, 0, stream>>>(Wk, wt + 1 * WSZ);
  cvt_wt<<<4096, 256, 0, stream>>>(Wv, wt + 2 * WSZ);
  cvt_wt<<<4096, 256, 0, stream>>>(Wu, wt + 3 * WSZ);
  cvt_wt<<<4096, 256, 0, stream>>>(Wo, wt + 4 * WSZ);

  dim3 gg(D_ / 128, BN / 128);  // (8, 256)
  gemm128<<<gg, 256, 0, stream>>>(qry_bf, wt + 0 * WSZ, bq, q_bf, BN, D_, D_, 0);
  gemm128<<<gg, 256, 0, stream>>>(qry_bf, wt + 1 * WSZ, bk, kT,   BN, D_, D_, 1);
  gemm128<<<gg, 256, 0, stream>>>(qry_bf, wt + 2 * WSZ, bv, vT,   BN, D_, D_, 2);
  gemm128<<<gg, 256, 0, stream>>>(qry_bf, wt + 3 * WSZ, bu, u,    BN, D_, D_, 3);

  kv_partial<<<dim3(64, 8), 256, 0, stream>>>(kT, vT, kvp);
  kv_reduce<<<(64 * 4096) / 256, 256, 0, stream>>>(kvp, kvT);
  attn_out<<<dim3(NSEQ / 128, 64), 256, 0, stream>>>(q_bf, kvT, attn);
  ln_mul<<<BN, 256, 0, stream>>>(attn, u, lnw, lnb, x2);

  gemm128<<<gg, 256, 0, stream>>>(x2, wt + 4 * WSZ, bo, d_out, BN, D_, D_, 3);
}
